// SignatureLSTM_89550068122115
// MI455X (gfx1250) — matrix operand order repaired and checked by reading, not compiled, not run on hardware
//
#include <hip/hip_runtime.h>
#include <stdint.h>

// Problem constants (match reference)
#define B_  64
#define T_  512
#define D_  32
#define U_  512
#define S_  1056          // D + D*D
#define BT_ (B_*T_)

typedef __bf16 v16bf __attribute__((ext_vector_type(16)));
typedef float  v8f   __attribute__((ext_vector_type(8)));

union ABFrag { uint4 q[2]; v16bf v; };

static __device__ __forceinline__ unsigned short f2bf(float f) {
  unsigned int u = __float_as_uint(f);
  u += 0x7FFFu + ((u >> 16) & 1u);          // round-to-nearest-even
  return (unsigned short)(u >> 16);
}

static __device__ __forceinline__ float sigmoidf_(float x) {
  return 1.0f / (1.0f + __expf(-x));
}

// A-fragment: 16x32 bf16, row-major source, stride in elements.
// lanes 0-15: M=lane, K = k0+0..7 (regs0-3), k0+16..23 (regs4-7)
// lanes16-31: M=lane-16, K = k0+8..15, k0+24..31
static __device__ __forceinline__ void load_A(const unsigned short* base, int stride,
                                              int m0, int k0, int lane, ABFrag& a) {
  const unsigned short* p = base + (size_t)(m0 + (lane & 15)) * stride
                                 + k0 + ((lane & 16) ? 8 : 0);
  a.q[0] = *reinterpret_cast<const uint4*>(p);
  a.q[1] = *reinterpret_cast<const uint4*>(p + 16);
}

// B-fragment: 32x16 bf16 from transposed (N-major) weights, stride in elements.
// lanes 0-15: N=lane, K = k0+0..7, k0+16..23; lanes16-31: N=lane-16, K = k0+8..15, k0+24..31
static __device__ __forceinline__ void load_B(const unsigned short* base, int stride,
                                              int n0, int k0, int lane, ABFrag& b) {
  const unsigned short* p = base + (size_t)(n0 + (lane & 15)) * stride
                                 + k0 + ((lane & 16) ? 8 : 0);
  b.q[0] = *reinterpret_cast<const uint4*>(p);
  b.q[1] = *reinterpret_cast<const uint4*>(p + 16);
}

// ---------------------------------------------------------------------------
// Kernel 1: weight transpose + bf16 conversion + state init (one flat pass)
// ---------------------------------------------------------------------------
__global__ void prep_kernel(const float* __restrict__ x,  const float* __restrict__ ik,
                            const float* __restrict__ rk, const float* __restrict__ fk,
                            unsigned short* WfT, unsigned short* WrT, unsigned short* IkT,
                            unsigned short* xbf, unsigned short* hbf0, float* c) {
  int idx = blockIdx.x * 256 + threadIdx.x;
  const int R0 = S_ * U_;                    // WfT: (U x S) = fk^T
  const int R1 = R0 + 3 * U_ * U_;           // WrT: (3U x U) = rk^T
  const int R2 = R1 + 3 * U_ * D_;           // IkT: (3U x D) = ik^T
  const int R3 = R2 + BT_ * D_;              // xbf
  const int R4 = R3 + B_ * U_;               // h buffer 0 = 0
  const int R5 = R4 + B_ * U_;               // c = 0
  if (idx < R0) {
    int u = idx / S_, s = idx % S_;
    WfT[idx] = f2bf(fk[(size_t)s * U_ + u]);
  } else if (idx < R1) {
    int i2 = idx - R0; int j = i2 / U_, k = i2 % U_;
    WrT[i2] = f2bf(rk[(size_t)k * (3*U_) + j]);
  } else if (idx < R2) {
    int i3 = idx - R1; int j = i3 / D_, d = i3 % D_;
    IkT[i3] = f2bf(ik[(size_t)d * (3*U_) + j]);
  } else if (idx < R3) {
    int i4 = idx - R2;
    xbf[i4] = f2bf(x[i4]);
  } else if (idx < R4) {
    hbf0[idx - R3] = 0;
  } else if (idx < R5) {
    c[idx - R4] = 0.0f;
  }
}

// ---------------------------------------------------------------------------
// Kernel 2: depth-2 streamed path signature (Chen), normalized, bf16 output.
// One block per batch element; S1/S2 state lives in LDS.
// ---------------------------------------------------------------------------
__global__ void sig_kernel(const float* __restrict__ x, unsigned short* __restrict__ ns) {
  __shared__ float S1s[32];
  __shared__ float S2s[1024];
  __shared__ float dxs[32];
  int b = blockIdx.x, tid = threadIdx.x;
  if (tid < 32) S1s[tid] = 0.0f;
  for (int i = tid; i < 1024; i += 256) S2s[i] = 0.0f;
  size_t row0 = (size_t)b * T_ * S_;
  for (int i = tid; i < S_; i += 256) ns[row0 + i] = 0;   // norm_sigs[:,0,:] = 0
  __syncthreads();
  const float* xb = x + (size_t)b * T_ * D_;
  for (int t = 1; t < T_; ++t) {
    if (tid < 32) dxs[tid] = xb[t * D_ + tid] - xb[(t - 1) * D_ + tid];
    __syncthreads();
    for (int i = tid; i < 1024; i += 256) {
      int r = i >> 5, cc = i & 31;
      S2s[i] += S1s[r] * dxs[cc] + 0.5f * dxs[r] * dxs[cc];   // uses old S1
    }
    __syncthreads();
    if (tid < 32) S1s[tid] += dxs[tid];
    __syncthreads();
    float inv = 1.0f / (float)t;
    size_t row = row0 + (size_t)t * S_;
    for (int i = tid; i < S_; i += 256) {
      float v = (i < 32) ? S1s[i] : S2s[i - 32];
      ns[row + i] = f2bf(v * inv);
    }
    __syncthreads();
  }
}

// ---------------------------------------------------------------------------
// Kernel 3: x_proj = x @ input_kernel   (M=BT, N=1536, K=32; single K-step)
// Each wave: 16x64 tile (4 WMMA).
// ---------------------------------------------------------------------------
__global__ void gemm_xproj(const unsigned short* __restrict__ xbf,
                           const unsigned short* __restrict__ IkT,
                           float* __restrict__ xproj) {
  int lane = threadIdx.x & 31, wid = threadIdx.x >> 5;
  int gw = blockIdx.x * 8 + wid;
  int mt = gw / 24, ng = gw % 24;
  int m0 = mt * 16, n0 = ng * 64;
  ABFrag a; load_A(xbf, D_, m0, 0, lane, a);
#pragma unroll
  for (int s = 0; s < 4; ++s) {
    ABFrag bb; load_B(IkT, D_, n0 + s * 16, 0, lane, bb);
    v8f z = {};
    v8f acc = __builtin_amdgcn_wmma_f32_16x16x32_bf16(false, a.v, false, bb.v,
                                                      (short)0, z, false, false);
    int n = n0 + s * 16 + (lane & 15);
#pragma unroll
    for (int r = 0; r < 8; ++r) {
      int m = m0 + r + 8 * (lane >> 4);
      xproj[(size_t)m * (3*U_) + n] = acc[r];
    }
  }
}

// ---------------------------------------------------------------------------
// Kernel 4: f_all = sigmoid(norm_sigs @ forget_kernel + b_f)
// M=BT, N=512, K=1056 (=33 k-steps). Each wave: 16x64 (4 acc, 4 WMMA/k-step).
// ---------------------------------------------------------------------------
__global__ void gemm_forget(const unsigned short* __restrict__ ns,
                            const unsigned short* __restrict__ WfT,
                            const float* __restrict__ bias,
                            float* __restrict__ fall) {
  int lane = threadIdx.x & 31, wid = threadIdx.x >> 5;
  int gw = blockIdx.x * 8 + wid;
  int mt = gw >> 3, ng = gw & 7;
  int m0 = mt * 16, n0 = ng * 64;
  v8f acc0 = {}, acc1 = {}, acc2 = {}, acc3 = {};
  for (int k0 = 0; k0 < S_; k0 += 32) {
    ABFrag a; load_A(ns, S_, m0, k0, lane, a);
    ABFrag b0; load_B(WfT, S_, n0 +  0, k0, lane, b0);
    ABFrag b1; load_B(WfT, S_, n0 + 16, k0, lane, b1);
    ABFrag b2; load_B(WfT, S_, n0 + 32, k0, lane, b2);
    ABFrag b3; load_B(WfT, S_, n0 + 48, k0, lane, b3);
    acc0 = __builtin_amdgcn_wmma_f32_16x16x32_bf16(false, a.v, false, b0.v, (short)0, acc0, false, false);
    acc1 = __builtin_amdgcn_wmma_f32_16x16x32_bf16(false, a.v, false, b1.v, (short)0, acc1, false, false);
    acc2 = __builtin_amdgcn_wmma_f32_16x16x32_bf16(false, a.v, false, b2.v, (short)0, acc2, false, false);
    acc3 = __builtin_amdgcn_wmma_f32_16x16x32_bf16(false, a.v, false, b3.v, (short)0, acc3, false, false);
  }
  v8f accs[4] = {acc0, acc1, acc2, acc3};
#pragma unroll
  for (int s = 0; s < 4; ++s) {
    int n = n0 + s * 16 + (lane & 15);
    float bf = bias[U_ + n];                 // b_f
#pragma unroll
    for (int r = 0; r < 8; ++r) {
      int m = m0 + r + 8 * (lane >> 4);
      fall[(size_t)m * U_ + n] = sigmoidf_(accs[s][r] + bf);
    }
  }
}

// ---------------------------------------------------------------------------
// Kernel 5: one fused LSTM step.
// gates = h @ Wr + x_proj[t]; pointwise i/c_hat/o + c,h update.
// 128 waves: 4 M-tiles x 32 N-tiles; each wave: 3 gate accumulators,
// 16 k-steps x 3 WMMA. h ping-pong in bf16 for the next step's A-operand.
// ---------------------------------------------------------------------------
__global__ void lstm_step(const unsigned short* __restrict__ hin,
                          const unsigned short* __restrict__ WrT,
                          const float* __restrict__ xproj,
                          const float* __restrict__ fall,
                          const float* __restrict__ bias,
                          unsigned short* __restrict__ hout,
                          float* __restrict__ c,
                          float* __restrict__ out, int t) {
  int lane = threadIdx.x & 31, wid = threadIdx.x >> 5;
  int gw = blockIdx.x * 8 + wid;
  int mt = gw >> 5, nt = gw & 31;
  int m0 = mt * 16, n0 = nt * 16;
  v8f gi_a = {}, gc_a = {}, go_a = {};
  for (int k0 = 0; k0 < U_; k0 += 32) {
    ABFrag a; load_A(hin, U_, m0, k0, lane, a);
    ABFrag b0; load_B(WrT, U_, 0 * U_ + n0, k0, lane, b0);
    ABFrag b1; load_B(WrT, U_, 1 * U_ + n0, k0, lane, b1);
    ABFrag b2; load_B(WrT, U_, 2 * U_ + n0, k0, lane, b2);
    gi_a = __builtin_amdgcn_wmma_f32_16x16x32_bf16(false, a.v, false, b0.v, (short)0, gi_a, false, false);
    gc_a = __builtin_amdgcn_wmma_f32_16x16x32_bf16(false, a.v, false, b1.v, (short)0, gc_a, false, false);
    go_a = __builtin_amdgcn_wmma_f32_16x16x32_bf16(false, a.v, false, b2.v, (short)0, go_a, false, false);
  }
  int n = n0 + (lane & 15);
  float bi = bias[n], bc = bias[2 * U_ + n], bo = bias[3 * U_ + n];
#pragma unroll
  for (int r = 0; r < 8; ++r) {
    int m = m0 + r + 8 * (lane >> 4);         // batch index
    size_t row = (size_t)m * T_ + t;
    float gi = gi_a[r] + xproj[row * (3*U_) +          n] + bi;
    float gc = gc_a[r] + xproj[row * (3*U_) +    U_ + n] + bc;
    float go = go_a[r] + xproj[row * (3*U_) + 2*U_ + n] + bo;
    float i_t = sigmoidf_(gi);
    float chat = tanhf(gc);
    float o_t = sigmoidf_(go);
    float f_t = fall[row * U_ + n];
    size_t hc = (size_t)m * U_ + n;
    float cn = f_t * c[hc] + i_t * chat;
    c[hc] = cn;
    float h = o_t * tanhf(cn);
    hout[hc] = f2bf(h);
    out[hc] = h;                               // last step's write is the answer
  }
}

// ---------------------------------------------------------------------------
// Host-side orchestration (graph-capture safe: only kernel launches on stream)
// ---------------------------------------------------------------------------
extern "C" void kernel_launch(void* const* d_in, const int* in_sizes, int n_in,
                              void* d_out, int out_size, void* d_ws, size_t ws_size,
                              hipStream_t stream) {
  (void)in_sizes; (void)n_in; (void)out_size; (void)ws_size;
  const float* x    = (const float*)d_in[0];
  const float* ik   = (const float*)d_in[1];
  const float* rk   = (const float*)d_in[2];
  const float* fk   = (const float*)d_in[3];
  const float* bias = (const float*)d_in[4];
  float* out = (float*)d_out;

  char* ws = (char*)d_ws;
  // byte offsets (all 256B-aligned)
  unsigned short* ns   = (unsigned short*)(ws);               // 32768*1056 bf16 = 69,206,016 B
  unsigned short* WfT  = (unsigned short*)(ws +  69206016);   //   512*1056 bf16 =  1,081,344 B
  unsigned short* WrT  = (unsigned short*)(ws +  70287360);   //  1536*512  bf16 =  1,572,864 B
  unsigned short* IkT  = (unsigned short*)(ws +  71860224);   //  1536*32   bf16 =     98,304 B
  unsigned short* xbf  = (unsigned short*)(ws +  71958528);   // 32768*32   bf16 =  2,097,152 B
  float*          xprj = (float*)         (ws +  74055680);   // 32768*1536 f32  = 201,326,592 B
  float*          fall = (float*)         (ws + 275382272);   // 32768*512  f32  =  67,108,864 B
  unsigned short* hbf  = (unsigned short*)(ws + 342491136);   // 2 x 64*512 bf16 =    131,072 B
  float*          cbuf = (float*)         (ws + 342622208);   //     64*512 f32  =    131,072 B

  prep_kernel<<<9728, 256, 0, stream>>>(x, ik, rk, fk, WfT, WrT, IkT, xbf, hbf, cbuf);
  sig_kernel<<<B_, 256, 0, stream>>>(x, ns);
  gemm_xproj<<<6144, 256, 0, stream>>>(xbf, IkT, xprj);      // 2048 m-tiles * 24 n-groups / 8 waves
  gemm_forget<<<2048, 256, 0, stream>>>(ns, WfT, bias, fall); // 2048 m-tiles * 8 n-groups / 8 waves

  for (int t = 0; t < T_; ++t) {
    unsigned short* hin  = hbf + (size_t)( t      & 1) * (B_ * U_);
    unsigned short* hout = hbf + (size_t)((t + 1) & 1) * (B_ * U_);
    lstm_step<<<16, 256, 0, stream>>>(hin, WrT, xprj, fall, bias, hout, cbuf, out, t);
  }
}
